// CommitRankingModule_70042326663858
// MI455X (gfx1250) — matrix operand order repaired and checked by reading, not compiled, not run on hardware
//
#include <hip/hip_runtime.h>
#include <hip/hip_bf16.h>
#include <math.h>

typedef __attribute__((ext_vector_type(16))) __bf16 v16bf;
typedef __attribute__((ext_vector_type(8)))  float  v8f;

// ---------- helpers ----------
__device__ __forceinline__ unsigned pack_bf16x2(float lo, float hi) {
  unsigned ul = __float_as_uint(lo);
  unsigned uh = __float_as_uint(hi);
  ul = ul + 0x7FFFu + ((ul >> 16) & 1u);   // RNE
  uh = uh + 0x7FFFu + ((uh >> 16) & 1u);
  return (ul >> 16) | (uh & 0xFFFF0000u);
}
__device__ __forceinline__ float gelu_exact(float v) {
  return 0.5f * v * (1.0f + erff(v * 0.70710678118654752f));
}
__device__ __forceinline__ unsigned fkey(float x) {   // monotonic float->uint
  unsigned u = __float_as_uint(x);
  return (u & 0x80000000u) ? ~u : (u | 0x80000000u);
}

// ---------- generic WMMA GEMM: C = act(alpha * A@B(+bias)) ----------
// A: [M,K] row-major fp32 (lda). B: opB=0 -> [K,N] row-major (ldb);
// opB=1 -> B element (k,n) = Bmem[n*ldb + k]  (A @ B^T).
// Block tile 128x64, 8 wave32s, each wave 32x32 (2x2 WMMA 16x16x32 bf16).
// LDS holds tiles pre-swizzled in WMMA fragment order:
//   lsA[tileM][lane][slot]  (uint = 2 packed bf16 = one A-frag VGPR)
//   lsB[tileN][lane][slot]
#define BM 128
#define BN 64
#define KT 32

__global__ __launch_bounds__(256) void gemm_wmma(
    const float* __restrict__ A, int lda,
    const float* __restrict__ B, int ldb,
    float* __restrict__ C, int ldc,
    const float* __restrict__ bias,
    int M, int N, int K, int opB, int act, float alpha)
{
  __shared__ unsigned lsA[BM / 16][32][8];   // 8 KB
  __shared__ unsigned lsB[BN / 16][32][8];   // 4 KB

  const int tid  = threadIdx.x;
  const int wave = tid >> 5;
  const int lane = tid & 31;
  const int wm   = wave >> 1;     // 0..3 -> 32-row strip
  const int wn   = wave & 1;      // 0..1 -> 32-col strip
  const int m0   = blockIdx.x * BM;
  const int n0   = blockIdx.y * BN;
  const int l15  = lane & 15;
  const int half = lane >> 4;

  v8f acc[2][2] = {};
  const int nk = (K + KT - 1) / KT;

  for (int kt = 0; kt < nk; ++kt) {
    const int k0 = kt * KT;
    const bool Afast = (m0 + BM <= M) && (k0 + KT <= K);
    const bool Bfast = (n0 + BN <= N) && (k0 + KT <= K);

    // prefetch next K tile of A (gfx1250 global_prefetch_b8)
    if (Afast && (k0 + 2 * KT <= K)) {
      __builtin_prefetch((const void*)(A + (size_t)(m0 + (tid & 127)) * lda + k0 + KT), 0, 0);
    }

    // ---- stage A: 128x32, packed bf16 pairs in fragment order ----
    if (Afast) {
      for (int p = tid; p < BM * 16; p += 256) {
        const int r = p >> 4, cp = p & 15;
        const float2 f = *(const float2*)(A + (size_t)(m0 + r) * lda + k0 + 2 * cp);
        const int h = cp >> 3;
        const int j = cp & 7;
        lsA[r >> 4][h * 16 + (r & 15)][j] = pack_bf16x2(f.x, f.y);
      }
    } else {
      for (int p = tid; p < BM * 16; p += 256) {
        const int r = p >> 4, cp = p & 15;
        const int gm = m0 + r, gk = k0 + 2 * cp;
        const float lo = (gm < M && gk     < K) ? A[(size_t)gm * lda + gk]     : 0.0f;
        const float hi = (gm < M && gk + 1 < K) ? A[(size_t)gm * lda + gk + 1] : 0.0f;
        const int h = cp >> 3;
        const int j = cp & 7;
        lsA[r >> 4][h * 16 + (r & 15)][j] = pack_bf16x2(lo, hi);
      }
    }

    // ---- stage B: 32x64, packed bf16 pairs (along K) in fragment order ----
    if (opB == 0) {
      if (Bfast) {
        for (int p = tid; p < 16 * BN; p += 256) {
          const int n = p & 63, kp = p >> 6;
          const float* bp = B + (size_t)(k0 + 2 * kp) * ldb + n0 + n;
          lsB[n >> 4][(kp >> 3) * 16 + (n & 15)][kp & 7] = pack_bf16x2(bp[0], bp[ldb]);
        }
      } else {
        for (int p = tid; p < 16 * BN; p += 256) {
          const int n = p & 63, kp = p >> 6;
          const int gn = n0 + n, gk = k0 + 2 * kp;
          const float lo = (gn < N && gk     < K) ? B[(size_t)gk * ldb + gn]       : 0.0f;
          const float hi = (gn < N && gk + 1 < K) ? B[(size_t)(gk + 1) * ldb + gn] : 0.0f;
          lsB[n >> 4][(kp >> 3) * 16 + (n & 15)][kp & 7] = pack_bf16x2(lo, hi);
        }
      }
    } else {
      if (Bfast) {
        for (int p = tid; p < 16 * BN; p += 256) {
          const int n = p & 63, kp = p >> 6;
          const float2 f = *(const float2*)(B + (size_t)(n0 + n) * ldb + k0 + 2 * kp);
          lsB[n >> 4][(kp >> 3) * 16 + (n & 15)][kp & 7] = pack_bf16x2(f.x, f.y);
        }
      } else {
        for (int p = tid; p < 16 * BN; p += 256) {
          const int n = p & 63, kp = p >> 6;
          const int gn = n0 + n, gk = k0 + 2 * kp;
          const float lo = (gn < N && gk     < K) ? B[(size_t)gn * ldb + gk]     : 0.0f;
          const float hi = (gn < N && gk + 1 < K) ? B[(size_t)gn * ldb + gk + 1] : 0.0f;
          lsB[n >> 4][(kp >> 3) * 16 + (n & 15)][kp & 7] = pack_bf16x2(lo, hi);
        }
      }
    }
    __syncthreads();

    // ---- fragments: each lane reads its 32 contiguous bytes (2x ds_load_b128)
    v16bf afr[2], bfr[2];
    #pragma unroll
    for (int i = 0; i < 2; ++i) {
      union { uint4 q[2]; v16bf v; } u;
      const uint4* pa = (const uint4*)&lsA[wm * 2 + i][lane][0];
      u.q[0] = pa[0]; u.q[1] = pa[1];
      afr[i] = u.v;
    }
    #pragma unroll
    for (int j = 0; j < 2; ++j) {
      union { uint4 q[2]; v16bf v; } u;
      const uint4* pb = (const uint4*)&lsB[wn * 2 + j][lane][0];
      u.q[0] = pb[0]; u.q[1] = pb[1];
      bfr[j] = u.v;
    }
    #pragma unroll
    for (int i = 0; i < 2; ++i)
      #pragma unroll
      for (int j = 0; j < 2; ++j)
        acc[i][j] = __builtin_amdgcn_wmma_f32_16x16x32_bf16(
            false, afr[i], false, bfr[j], (short)0, acc[i][j], false, false);
    __syncthreads();
  }

  // ---- epilogue ----
  #pragma unroll
  for (int i = 0; i < 2; ++i) {
    #pragma unroll
    for (int j = 0; j < 2; ++j) {
      const int gn = n0 + wn * 32 + j * 16 + l15;
      #pragma unroll
      for (int e = 0; e < 8; ++e) {
        const int gm = m0 + wm * 32 + i * 16 + half * 8 + e;
        if (gm < M && gn < N) {
          float v = acc[i][j][e] * alpha;
          if (bias) v += bias[gn];
          if (act == 1) v = gelu_exact(v);
          C[(size_t)gm * ldc + gn] = v;
        }
      }
    }
  }
}

// ---------- row-wise LayerNorm (+optional residual, +optional GELU) ----------
__global__ __launch_bounds__(256) void ln_kernel(
    const float* __restrict__ X, const float* __restrict__ R,
    const float* __restrict__ g, const float* __restrict__ b,
    float* __restrict__ Y, int cols, int do_gelu)
{
  __shared__ float red[256];
  const int row = blockIdx.x, t = threadIdx.x;
  float v = X[(size_t)row * cols + t];
  if (R) v += R[(size_t)row * cols + t];
  red[t] = v; __syncthreads();
  for (int s = 128; s > 0; s >>= 1) { if (t < s) red[t] += red[t + s]; __syncthreads(); }
  float mu = red[0] / (float)cols; __syncthreads();
  float d = v - mu;
  red[t] = d * d; __syncthreads();
  for (int s = 128; s > 0; s >>= 1) { if (t < s) red[t] += red[t + s]; __syncthreads(); }
  float var = red[0] / (float)cols;
  float o = d * rsqrtf(var + 1e-5f) * g[t] + b[t];
  if (do_gelu) o = gelu_exact(o);
  Y[(size_t)row * cols + t] = o;
}

// ---------- row-wise softmax, in place ----------
__global__ __launch_bounds__(256) void softmax_rows(float* __restrict__ S, int L)
{
  __shared__ float red[256];
  float* p = S + (size_t)blockIdx.x * L;
  const int t = threadIdx.x;
  float mx = -3.4e38f;
  for (int i = t; i < L; i += 256) mx = fmaxf(mx, p[i]);
  red[t] = mx; __syncthreads();
  for (int s = 128; s > 0; s >>= 1) { if (t < s) red[t] = fmaxf(red[t], red[t + s]); __syncthreads(); }
  mx = red[0]; __syncthreads();
  float sum = 0.0f;
  for (int i = t; i < L; i += 256) sum += expf(p[i] - mx);
  red[t] = sum; __syncthreads();
  for (int s = 128; s > 0; s >>= 1) { if (t < s) red[t] += red[t + s]; __syncthreads(); }
  const float inv = 1.0f / red[0];
  for (int i = t; i < L; i += 256) p[i] = expf(p[i] - mx) * inv;
}

// ---------- pooling: fold q into Wk:  Wkq[j,h] = sum_d Wk[j,h*64+d]*q[h,d] ----------
__global__ void build_combo(const float* __restrict__ Wk, const float* __restrict__ bk,
                            const float* __restrict__ q,
                            float* __restrict__ Wkq, float* __restrict__ bq,
                            int Hd, int H, int D)
{
  const int idx = blockIdx.x * blockDim.x + threadIdx.x;
  if (idx < Hd * H) {
    int j = idx / H, h = idx % H;
    float s = 0.0f;
    for (int d = 0; d < D; ++d) s += Wk[(size_t)j * Hd + h * D + d] * q[h * D + d];
    Wkq[j * H + h] = s;
  }
  if (idx < H) {
    float s = 0.0f;
    for (int d = 0; d < D; ++d) s += bk[idx * D + d] * q[idx * D + d];
    bq[idx] = s;
  }
}

// ---------- pooling pass 1: logits; segment max via key atomics (wave/node, H=4) ----------
__global__ __launch_bounds__(256) void pool_logits(
    const float* __restrict__ x, const float* __restrict__ Wkq,
    const float* __restrict__ bq, const int* __restrict__ ci,
    float* __restrict__ logits, unsigned* __restrict__ mkey,
    int N, int Hd, float scale)
{
  const int n = blockIdx.x * 8 + (threadIdx.x >> 5);
  const int lane = threadIdx.x & 31;
  if (n >= N) return;
  const float* xr = x + (size_t)n * Hd;
  float a0 = 0, a1 = 0, a2 = 0, a3 = 0;
  for (int j = lane; j < Hd; j += 32) {
    float xv = xr[j];
    a0 += xv * Wkq[j * 4 + 0];
    a1 += xv * Wkq[j * 4 + 1];
    a2 += xv * Wkq[j * 4 + 2];
    a3 += xv * Wkq[j * 4 + 3];
  }
  for (int m = 16; m > 0; m >>= 1) {
    a0 += __shfl_xor(a0, m, 32); a1 += __shfl_xor(a1, m, 32);
    a2 += __shfl_xor(a2, m, 32); a3 += __shfl_xor(a3, m, 32);
  }
  if (lane == 0) {
    const int c = ci[n];
    float lg[4] = { (a0 + bq[0]) * scale, (a1 + bq[1]) * scale,
                    (a2 + bq[2]) * scale, (a3 + bq[3]) * scale };
    for (int h = 0; h < 4; ++h) {
      logits[(size_t)n * 4 + h] = lg[h];
      atomicMax(&mkey[c * 4 + h], fkey(lg[h]));
    }
  }
}

__global__ void pool_decode_max(const unsigned* __restrict__ mkey,
                                float* __restrict__ mval, int CH)
{
  const int i = blockIdx.x * blockDim.x + threadIdx.x;
  if (i >= CH) return;
  unsigned k = mkey[i];
  float m = 0.0f;
  if (k != 0u) {
    unsigned u = (k & 0x80000000u) ? (k & 0x7FFFFFFFu) : ~k;
    m = __uint_as_float(u);
    if (!isfinite(m)) m = 0.0f;
  }
  mval[i] = m;
}

// ---------- pooling pass 2: e=exp(logit-m); s += e; pooledAcc += e*V ----------
__global__ void pool_accum(const float* __restrict__ logits, const float* __restrict__ mval,
                           const float* __restrict__ V, const int* __restrict__ ci,
                           float* __restrict__ s, float* __restrict__ pooledAcc,
                           int N, int Hd)
{
  const int n = blockIdx.x * blockDim.x + threadIdx.x;
  if (n >= N) return;
  const int c = ci[n];
  for (int h = 0; h < 4; ++h) {
    float e = expf(logits[(size_t)n * 4 + h] - mval[c * 4 + h]);
    atomicAdd(&s[c * 4 + h], e);
    const float* vr = V + (size_t)n * Hd + h * 64;
    float* pr = pooledAcc + (size_t)c * Hd + h * 64;
    for (int d = 0; d < 64; ++d) atomicAdd(&pr[d], e * vr[d]);
  }
}

__global__ void pool_final(const float* __restrict__ pooledAcc, const float* __restrict__ s,
                           float* __restrict__ pooled, int C, int Hd)
{
  const int i = blockIdx.x * blockDim.x + threadIdx.x;
  if (i >= C * Hd) return;
  const int c = i / Hd, h = (i % Hd) / 64;
  float ss = s[c * 4 + h];
  pooled[i] = (ss > 0.0f) ? pooledAcc[i] / ss : 0.0f;
}

// ---------- host ----------
static inline void launch_gemm(const float* A, int lda, const float* B, int ldb,
                               float* Cc, int ldc, const float* bias,
                               int M, int N, int K, int opB, int act, float alpha,
                               hipStream_t st)
{
  dim3 grid((M + BM - 1) / BM, (N + BN - 1) / BN);
  gemm_wmma<<<grid, 256, 0, st>>>(A, lda, B, ldb, Cc, ldc, bias, M, N, K, opB, act, alpha);
}

extern "C" void kernel_launch(void* const* d_in, const int* in_sizes, int n_in,
                              void* d_out, int out_size, void* d_ws, size_t ws_size,
                              hipStream_t stream)
{
  const int Hd = in_sizes[4];           // 256
  const int IN = in_sizes[3] / Hd;      // 768
  const int Nn = in_sizes[1];           // 300000
  const int FF = in_sizes[27];          // 1024
  const int C  = out_size;              // 2000
  const int D  = 64;
  const float scale = 0.125f;           // D^-0.5

  const float* node_emb = (const float*)d_in[0];
  const int*   ci       = (const int*)  d_in[1];
  const float* W_in = (const float*)d_in[3],  *b_in = (const float*)d_in[4];
  const float* g_in = (const float*)d_in[5],  *be_in = (const float*)d_in[6];
  const float* q_c  = (const float*)d_in[7];
  const float* Wk = (const float*)d_in[8],  *bk = (const float*)d_in[9];
  const float* Wv = (const float*)d_in[10], *bv = (const float*)d_in[11];
  const float* Wo_p = (const float*)d_in[12], *bo_p = (const float*)d_in[13];
  const float* g_p  = (const float*)d_in[14], *b_p  = (const float*)d_in[15];
  const float* Wq_t = (const float*)d_in[16], *bq_t = (const float*)d_in[17];
  const float* Wk_t = (const float*)d_in[18], *bk_t = (const float*)d_in[19];
  const float* Wv_t = (const float*)d_in[20], *bv_t = (const float*)d_in[21];
  const float* Wo_t = (const float*)d_in[22], *bo_t = (const float*)d_in[23];
  const float* g1 = (const float*)d_in[24], *b1 = (const float*)d_in[25];
  const float* Wf1 = (const float*)d_in[26], *bf1 = (const float*)d_in[27];
  const float* Wf2 = (const float*)d_in[28], *bf2 = (const float*)d_in[29];
  const float* g2 = (const float*)d_in[30], *b2 = (const float*)d_in[31];
  const float* Wh1 = (const float*)d_in[32], *bh1 = (const float*)d_in[33];
  const float* Wh2 = (const float*)d_in[34], *bh2 = (const float*)d_in[35];

  float* w = (float*)d_ws;
  size_t o = 0;
  float* x       = w + o; o += (size_t)Nn * Hd;
  float* Vmat    = w + o; o += (size_t)Nn * Hd;
  float* logits  = w + o; o += (size_t)Nn * 4;
  float* Wkq     = w + o; o += (size_t)Hd * 4;
  float* bqv     = w + o; o += 4;
  unsigned* mkey = (unsigned*)(w + o); o += (size_t)C * 4;
  float* mval    = w + o; o += (size_t)C * 4;
  float* ssum    = w + o; o += (size_t)C * 4;
  float* pAcc    = w + o; o += (size_t)C * Hd;
  float* pooled  = w + o; o += (size_t)C * Hd;
  float* emb     = w + o; o += (size_t)C * Hd;
  float* qb      = w + o; o += (size_t)C * Hd;
  float* kb      = w + o; o += (size_t)C * Hd;
  float* vb      = w + o; o += (size_t)C * Hd;
  float* sc      = w + o; o += (size_t)4 * C * C;
  float* att     = w + o; o += (size_t)C * Hd;
  float* saPr    = w + o; o += (size_t)C * Hd;
  float* y       = w + o; o += (size_t)C * Hd;
  float* ffh     = w + o; o += (size_t)C * FF;
  float* ffo     = w + o; o += (size_t)C * Hd;
  float* y2      = w + o; o += (size_t)C * Hd;
  float* h1b     = w + o; o += (size_t)C * (Hd / 2);

  hipMemsetAsync(mkey, 0, (size_t)C * 4 * sizeof(unsigned), stream);
  hipMemsetAsync(ssum, 0, (size_t)C * 4 * sizeof(float), stream);
  hipMemsetAsync(pAcc, 0, (size_t)C * Hd * sizeof(float), stream);

  // 1) input_proj GEMM -> x, then LN+GELU in place
  launch_gemm(node_emb, IN, W_in, Hd, x, Hd, b_in, Nn, Hd, IN, 0, 0, 1.0f, stream);
  ln_kernel<<<Nn, Hd, 0, stream>>>(x, nullptr, g_in, be_in, x, Hd, 1);

  // 2) V projection (WMMA)
  launch_gemm(x, Hd, Wv, Hd, Vmat, Hd, bv, Nn, Hd, Hd, 0, 0, 1.0f, stream);

  // 3) pooling: logits via folded Wk@q, segment softmax with atomics
  build_combo<<<(Hd * 4 + 255) / 256, 256, 0, stream>>>(Wk, bk, q_c, Wkq, bqv, Hd, 4, D);
  pool_logits<<<(Nn + 7) / 8, 256, 0, stream>>>(x, Wkq, bqv, ci, logits, mkey, Nn, Hd, scale);
  pool_decode_max<<<(C * 4 + 255) / 256, 256, 0, stream>>>(mkey, mval, C * 4);
  pool_accum<<<(Nn + 255) / 256, 256, 0, stream>>>(logits, mval, Vmat, ci, ssum, pAcc, Nn, Hd);
  pool_final<<<((C * Hd) + 255) / 256, 256, 0, stream>>>(pAcc, ssum, pooled, C, Hd);

  // 4) pooled projection + LN -> emb
  launch_gemm(pooled, Hd, Wo_p, Hd, emb, Hd, bo_p, C, Hd, Hd, 0, 0, 1.0f, stream);
  ln_kernel<<<C, Hd, 0, stream>>>(emb, nullptr, g_p, b_p, emb, Hd, 0);

  // 5) transformer encoder layer over commits
  launch_gemm(emb, Hd, Wq_t, Hd, qb, Hd, bq_t, C, Hd, Hd, 0, 0, 1.0f, stream);
  launch_gemm(emb, Hd, Wk_t, Hd, kb, Hd, bk_t, C, Hd, Hd, 0, 0, 1.0f, stream);
  launch_gemm(emb, Hd, Wv_t, Hd, vb, Hd, bv_t, C, Hd, Hd, 0, 0, 1.0f, stream);
  for (int h = 0; h < 4; ++h)     // scores = scale * q_h @ k_h^T
    launch_gemm(qb + h * D, Hd, kb + h * D, Hd, sc + (size_t)h * C * C, C,
                nullptr, C, C, D, 1, 0, scale, stream);
  softmax_rows<<<4 * C, 256, 0, stream>>>(sc, C);
  for (int h = 0; h < 4; ++h)     // att_h = a_h @ v_h
    launch_gemm(sc + (size_t)h * C * C, C, vb + h * D, Hd, att + h * D, Hd,
                nullptr, C, D, C, 0, 0, 1.0f, stream);
  launch_gemm(att, Hd, Wo_t, Hd, saPr, Hd, bo_t, C, Hd, Hd, 0, 0, 1.0f, stream);
  ln_kernel<<<C, Hd, 0, stream>>>(emb, saPr, g1, b1, y, Hd, 0);
  launch_gemm(y, Hd, Wf1, FF, ffh, FF, bf1, C, FF, Hd, 0, 1, 1.0f, stream);
  launch_gemm(ffh, FF, Wf2, Hd, ffo, Hd, bf2, C, Hd, FF, 0, 0, 1.0f, stream);
  ln_kernel<<<C, Hd, 0, stream>>>(y, ffo, g2, b2, y2, Hd, 0);

  // 6) ranking head -> d_out [C]
  launch_gemm(y2, Hd, Wh1, Hd / 2, h1b, Hd / 2, bh1, C, Hd / 2, Hd, 0, 1, 1.0f, stream);
  launch_gemm(h1b, Hd / 2, Wh2, 1, (float*)d_out, 1, bh2, C, 1, Hd / 2, 0, 0, 1.0f, stream);
  (void)n_in; (void)ws_size;
}
